// GINE_13898514170648
// MI455X (gfx1250) — matrix operand order repaired and checked by reading, not compiled, not run on hardware
//
#include <hip/hip_runtime.h>
#include <hip/hip_bf16.h>

#define H 256
#define BSTR 264  // padded LDS K-stride (bf16 elems): 264*2B=528B -> breaks 64-bank conflicts

typedef __attribute__((ext_vector_type(16))) __bf16 v16bf;
typedef __attribute__((ext_vector_type(8)))  float  v8f;
typedef __attribute__((ext_vector_type(4)))  unsigned v4u;
typedef __attribute__((ext_vector_type(8)))  int      v8i;
typedef __attribute__((ext_vector_type(4)))  int      v4i;

__device__ __forceinline__ v8f wmma_bf16(v16bf a, v16bf b, v8f c) {
  // D = A(16x32 bf16) * B(32x16 bf16) + C(f32)
  return __builtin_amdgcn_wmma_f32_16x16x32_bf16(false, a, false, b, (short)0, c, false, false);
}

// ---------------------------------------------------------------------------
// TDM: DMA a rows x 256 bf16 weight tile (row-major) from global into LDS,
// with hardware padding of 4 dwords every 128 dwords -> LDS row stride = BSTR.
// D# built per CDNA5 ISA 8.3/8.4. Issued by one wave; completion = TENSORcnt.
// ---------------------------------------------------------------------------
__device__ __forceinline__ void tdm_load_w(const __bf16* gsrc, __bf16* lds, int rows) {
  unsigned ldsaddr = (unsigned)(unsigned long long)lds;
  unsigned long long ga = (unsigned long long)gsrc;
  v4u g0;
  g0[0] = 1u;                                      // count=1 (valid user descriptor)
  g0[1] = ldsaddr;                                 // lds_addr (bytes)
  g0[2] = (unsigned)(ga & 0xFFFFFFFFu);            // global_addr[31:0]
  g0[3] = (unsigned)((ga >> 32) & 0x01FFFFFFu)     // global_addr[56:32]
        | 0x80000000u;                             // type=2 ("image")
  v8i g1;
  // dword0: data_size=1 (2B) | pad_enable | pad_interval=6 (128 dw) | pad_amount=3 (4 dw)
  g1[0] = (1 << 16) | (1 << 20) | (6 << 22) | (3 << 25);
  g1[1] = 0x01000000;                              // tensor_dim0=256 (bits 63:48)
  g1[2] = (int)((unsigned)rows << 16);             // tensor_dim1=rows (bits 95:80)
  g1[3] = 0x01000000;                              // tile_dim0=256 (bits 127:112)
  g1[4] = rows;                                    // tile_dim1=rows (bits 143:128)
  g1[5] = 0x00000100;                              // tensor_dim0_stride=256 (bits 191:160)
  g1[6] = 0;
  g1[7] = 0;
  v4i z4 = {0, 0, 0, 0};
#if defined(__clang_major__) && __clang_major__ >= 23
  v8i z8 = {0, 0, 0, 0, 0, 0, 0, 0};
  __builtin_amdgcn_tensor_load_to_lds(g0, g1, z4, z4, z8, 0);
#else
  __builtin_amdgcn_tensor_load_to_lds(g0, g1, z4, z4, 0);
#endif
}

// A fragment: 16x32 bf16 tile at tileBase (row-major, stride lda elems), K offset kbase.
// Layout: lanes 0-15 hold rows M=0..15 with K=0..15 contiguous;
// lanes 16-31 hold the same rows with K=16..31. Each half merges into 2x global_load_b128.
__device__ __forceinline__ v16bf load_a_frag(const __bf16* tileBase, int lda, int kbase, int lane) {
  const __bf16* rowp = tileBase + (size_t)(lane & 15) * lda + kbase + ((lane >> 4) << 4);
  v16bf a;
  unsigned* au = (unsigned*)&a;
#pragma unroll
  for (int i = 0; i < 8; ++i) {
    int k = 2 * i;
    au[i] = *(const unsigned*)(rowp + k);
  }
  return a;
}

// B fragment from LDS, layout Blds[n*BSTR + k] (i.e. W^T): lane holds column N=lane%16,
// K = kbase + (lane<16 ? 0..15 : 16..31) -> 8 contiguous dwords = 2x ds_load_b128.
__device__ __forceinline__ v16bf load_b_frag(const __bf16* Blds, int n0, int kbase, int lane) {
  const __bf16* p = Blds + (size_t)(n0 + (lane & 15)) * BSTR + kbase + ((lane >> 4) << 4);
  v16bf b;
  unsigned* bu = (unsigned*)&b;
  const unsigned* pu = (const unsigned*)p;
#pragma unroll
  for (int i = 0; i < 8; ++i) bu[i] = pu[i];
  return b;
}

// MODE bits: 1=relu, 2=eval-BN, 4=residual add, 8=write f32, 16=write bf16
template <int MODE>
__device__ __forceinline__ void gemm_epilogue(
    v8f c, int tileM, int mofs, int n, float bia, float scale, float shift,
    const float* __restrict__ resid, float* __restrict__ outF, __bf16* __restrict__ outB)
{
#pragma unroll
  for (int r = 0; r < 8; ++r) {
    int row = tileM * 16 + mofs + r;
    float v = c[r] + bia;
    if (MODE & 2) v = v * scale + shift;
    if (MODE & 1) v = fmaxf(v, 0.f);
    size_t o = (size_t)row * H + n;
    if (MODE & 4) v += resid[o];
    if (MODE & 8) outF[o] = v;
    if (MODE & 16) outB[o] = (__bf16)v;
  }
}

// ---------------------------------------------------------------------------
// Generic 256-wide GEMM: out = post(A[M,256](bf16) @ Bt[256,256](bf16,W^T) + bias)
// 8 waves/block, 2 x 16-row tiles per wave (B fragments reused for 2 WMMAs),
// weights DMA'd to LDS via TDM. Compile-time epilogue via MODE.
// ---------------------------------------------------------------------------
template <int MODE>
__global__ __launch_bounds__(256) void k_gemm(
    const __bf16* __restrict__ A, const __bf16* __restrict__ Bt,
    const float* __restrict__ bias,
    const float* __restrict__ bn_g, const float* __restrict__ bn_b,
    const float* __restrict__ resid,
    float* __restrict__ outF, __bf16* __restrict__ outB,
    int Mtiles)
{
  __shared__ __bf16 Blds[H * BSTR];
  if (threadIdx.x < 32) {                 // wave 0 issues the tensor DMA
    tdm_load_w(Bt, Blds, H);
    __builtin_amdgcn_s_wait_tensorcnt(0);
  }
  __syncthreads();

  int tid = threadIdx.x;
  int lane = tid & 31, wave = tid >> 5;
  int t0 = (blockIdx.x * 8 + wave) * 2;    // Mtiles is even for all our shapes
  if (t0 >= Mtiles) return;

  const __bf16* Ab0 = A + (size_t)t0 * 16 * H;
  const __bf16* Ab1 = Ab0 + (size_t)16 * H;
  v16bf a0[8], a1[8];
#pragma unroll
  for (int ks = 0; ks < 8; ++ks) {
    a0[ks] = load_a_frag(Ab0, H, ks * 32, lane);
    a1[ks] = load_a_frag(Ab1, H, ks * 32, lane);
  }

  int n_lo = lane & 15;
  int mofs = (lane >> 4) << 3;       // C/D layout: lanes>=16 hold M = r+8

  for (int nt = 0; nt < 16; ++nt) {
    v8f c0 = {0.f, 0.f, 0.f, 0.f, 0.f, 0.f, 0.f, 0.f};
    v8f c1 = {0.f, 0.f, 0.f, 0.f, 0.f, 0.f, 0.f, 0.f};
#pragma unroll
    for (int ks = 0; ks < 8; ++ks) {
      v16bf b = load_b_frag(Blds, nt * 16, ks * 32, lane);
      c0 = wmma_bf16(a0[ks], b, c0);
      c1 = wmma_bf16(a1[ks], b, c1);
    }
    int n = nt * 16 + n_lo;
    float bia = bias[n];
    float scale = 1.f, shift = 0.f;
    if (MODE & 2) { scale = bn_g[n] * 0.9999950000374996f; shift = bn_b[n]; }  // g/sqrt(1+eps)
    gemm_epilogue<MODE>(c0, t0,     mofs, n, bia, scale, shift, resid, outF, outB);
    gemm_epilogue<MODE>(c1, t0 + 1, mofs, n, bia, scale, shift, resid, outF, outB);
  }
}

// ---------------------------------------------------------------------------
// Fused edge projection + message + scatter (layers 1..4):
//   e = ea @ lw + lb ; m = relu(h[src] + e) ; agg[dst] += m
// WMMA for the GEMM, fp32 global atomics for segment_sum (agg fits in 192MB L2).
// ---------------------------------------------------------------------------
__global__ __launch_bounds__(256) void k_msg_scatter(
    const __bf16* __restrict__ ea, const __bf16* __restrict__ lwT,
    const float* __restrict__ lb, const float* __restrict__ h,
    const int* __restrict__ src, const int* __restrict__ dst,
    float* __restrict__ agg, int Etiles)
{
  __shared__ __bf16 Blds[H * BSTR];
  if (threadIdx.x < 32) {
    tdm_load_w(lwT, Blds, H);
    __builtin_amdgcn_s_wait_tensorcnt(0);
  }
  __syncthreads();

  int tid = threadIdx.x;
  int lane = tid & 31, wave = tid >> 5;
  int t0 = (blockIdx.x * 8 + wave) * 2;    // Etiles is even
  if (t0 >= Etiles) return;

  const __bf16* Ab0 = ea + (size_t)t0 * 16 * H;
  const __bf16* Ab1 = Ab0 + (size_t)16 * H;
  v16bf a0[8], a1[8];
#pragma unroll
  for (int ks = 0; ks < 8; ++ks) {
    a0[ks] = load_a_frag(Ab0, H, ks * 32, lane);
    a1[ks] = load_a_frag(Ab1, H, ks * 32, lane);
  }

  int n_lo = lane & 15;
  int mofs = (lane >> 4) << 3;
  int srcs[16], dsts[16];
#pragma unroll
  for (int r = 0; r < 8; ++r) {
    int e0 = t0 * 16 + mofs + r;
    int e1 = e0 + 16;
    srcs[r] = src[e0];     dsts[r] = dst[e0];
    srcs[8 + r] = src[e1]; dsts[8 + r] = dst[e1];
  }

  for (int nt = 0; nt < 16; ++nt) {
    v8f c0 = {0.f, 0.f, 0.f, 0.f, 0.f, 0.f, 0.f, 0.f};
    v8f c1 = {0.f, 0.f, 0.f, 0.f, 0.f, 0.f, 0.f, 0.f};
#pragma unroll
    for (int ks = 0; ks < 8; ++ks) {
      v16bf b = load_b_frag(Blds, nt * 16, ks * 32, lane);
      c0 = wmma_bf16(a0[ks], b, c0);
      c1 = wmma_bf16(a1[ks], b, c1);
    }
    int n = nt * 16 + n_lo;
    float bia = lb[n];
#pragma unroll
    for (int r = 0; r < 8; ++r) {
      float v = c0[r] + bia + h[(size_t)srcs[r] * H + n];
      atomicAdd(&agg[(size_t)dsts[r] * H + n], fmaxf(v, 0.f));
    }
#pragma unroll
    for (int r = 0; r < 8; ++r) {
      float v = c1[r] + bia + h[(size_t)srcs[8 + r] * H + n];
      atomicAdd(&agg[(size_t)dsts[8 + r] * H + n], fmaxf(v, 0.f));
    }
  }
}

// ---------------------------------------------------------------------------
// Layer-0 edge path as a 1-N-tile WMMA GEMM (out dim 9 padded to 16):
//   e9 = ea @ c0_lin_w(+pad) ; m = relu(x[src] + e9 + lb) ; agg9[dst] += m
// Reads ea exactly once (vs 9x in a scalar formulation).
// ---------------------------------------------------------------------------
__global__ __launch_bounds__(256) void k_msg_scatter9(
    const __bf16* __restrict__ ea, const __bf16* __restrict__ lwT9,
    const float* __restrict__ lb9, const float* __restrict__ x,
    const int* __restrict__ src, const int* __restrict__ dst,
    float* __restrict__ agg9, int Etiles)
{
  __shared__ __bf16 Blds[16 * BSTR];
  if (threadIdx.x < 32) {
    tdm_load_w(lwT9, Blds, 16);
    __builtin_amdgcn_s_wait_tensorcnt(0);
  }
  __syncthreads();

  int tid = threadIdx.x;
  int lane = tid & 31, wave = tid >> 5;
  int t0 = (blockIdx.x * 8 + wave) * 2;
  if (t0 >= Etiles) return;

  const __bf16* Ab0 = ea + (size_t)t0 * 16 * H;
  const __bf16* Ab1 = Ab0 + (size_t)16 * H;

  v8f c0 = {0.f, 0.f, 0.f, 0.f, 0.f, 0.f, 0.f, 0.f};
  v8f c1 = {0.f, 0.f, 0.f, 0.f, 0.f, 0.f, 0.f, 0.f};
#pragma unroll
  for (int ks = 0; ks < 8; ++ks) {
    v16bf b = load_b_frag(Blds, 0, ks * 32, lane);
    c0 = wmma_bf16(load_a_frag(Ab0, H, ks * 32, lane), b, c0);
    c1 = wmma_bf16(load_a_frag(Ab1, H, ks * 32, lane), b, c1);
  }

  int n = lane & 15;
  int mofs = (lane >> 4) << 3;
  if (n < 9) {
    float bia = lb9[n];
#pragma unroll
    for (int r = 0; r < 8; ++r) {
      int e0 = t0 * 16 + mofs + r;
      float v = c0[r] + bia + x[(size_t)src[e0] * 9 + n];
      atomicAdd(&agg9[(size_t)dst[e0] * 9 + n], fmaxf(v, 0.f));
      int e1 = e0 + 16;
      float w = c1[r] + bia + x[(size_t)src[e1] * 9 + n];
      atomicAdd(&agg9[(size_t)dst[e1] * 9 + n], fmaxf(w, 0.f));
    }
  }
}

// --------------------------- small scalar kernels ---------------------------

// W[K=256][N=256] (f32) -> Bt[n*256 + k] (bf16)  (transposed for B-fragment loads)
__global__ __launch_bounds__(256) void k_convT(const float* __restrict__ W, __bf16* __restrict__ Bt) {
  Bt[(size_t)blockIdx.x * H + threadIdx.x] = (__bf16)W[(size_t)threadIdx.x * H + blockIdx.x];
}

// c0_lin_w[K=256][9] (f32) -> Bt9[n*256 + k] (bf16), columns 9..15 zero-padded
__global__ __launch_bounds__(256) void k_convT9(const float* __restrict__ W, __bf16* __restrict__ Bt) {
  int n = blockIdx.x, k = threadIdx.x;
  float v = (n < 9) ? W[(size_t)k * 9 + n] : 0.f;
  Bt[(size_t)n * H + k] = (__bf16)v;
}

// hidden = relu(edge_attr @ em_w1 + em_b1)  [E,256] bf16 ; K=3
__global__ __launch_bounds__(256) void k_edge_hidden(
    const float* __restrict__ ea3, const float* __restrict__ w1,
    const float* __restrict__ b1, __bf16* __restrict__ out)
{
  int e = blockIdx.x, n = threadIdx.x;
  float a0 = ea3[3 * e], a1 = ea3[3 * e + 1], a2 = ea3[3 * e + 2];
  float v = fmaf(a0, w1[n], fmaf(a1, w1[H + n], fmaf(a2, w1[2 * H + n], b1[n])));
  out[(size_t)e * H + n] = (__bf16)fmaxf(v, 0.f);
}

// layer-0 node MLP stage 1 (K=9): t = relu((agg9+x) @ c0_w1 + c0_b1) -> bf16
__global__ __launch_bounds__(256) void k_c0_mlp1(
    const float* __restrict__ agg9, const float* __restrict__ x,
    const float* __restrict__ w, const float* __restrict__ b,
    __bf16* __restrict__ out, int N)
{
  int idx = blockIdx.x * blockDim.x + threadIdx.x;
  if (idx >= N * H) return;
  int nd = idx >> 8, n = idx & 255;
  float acc = b[n];
#pragma unroll
  for (int k = 0; k < 9; ++k)
    acc = fmaf(agg9[nd * 9 + k] + x[nd * 9 + k], w[k * H + n], acc);
  out[idx] = (__bf16)fmaxf(acc, 0.f);
}

// A = bf16(agg + h)
__global__ __launch_bounds__(256) void k_prep(
    const float* __restrict__ agg, const float* __restrict__ h, __bf16* __restrict__ A, int N)
{
  int idx = blockIdx.x * blockDim.x + threadIdx.x;
  if (idx >= N * H) return;
  A[idx] = (__bf16)(agg[idx] + h[idx]);
}

// per (node,ch): sums[b] += h ; mx[b] = max (h>=0 -> bit-monotone uint atomicMax)
__global__ __launch_bounds__(256) void k_pool_nh(
    const float* __restrict__ h, const int* __restrict__ batch,
    float* __restrict__ sums, float* __restrict__ mx, int N)
{
  int idx = blockIdx.x * blockDim.x + threadIdx.x;
  if (idx >= N * H) return;
  int nd = idx >> 8, n = idx & 255;
  int b = batch[nd];
  float v = h[idx];
  atomicAdd(&sums[(size_t)b * H + n], v);
  atomicMax((unsigned*)&mx[(size_t)b * H + n], __float_as_uint(v));
}

// gate = G1 @ gate_w2 + gate_b2 ; order-preserving-encoded atomicMax ; counts
__global__ __launch_bounds__(256) void k_gate2(
    const __bf16* __restrict__ G1, const float* __restrict__ gw2, const float* __restrict__ gb2,
    const int* __restrict__ batch, float* __restrict__ gate, unsigned* __restrict__ gmaxe,
    float* __restrict__ counts, int N)
{
  int nd = blockIdx.x * blockDim.x + threadIdx.x;
  if (nd >= N) return;
  const uint4* rp = (const uint4*)(G1 + (size_t)nd * H);  // 8 bf16 per uint4
  float acc = gb2[0];
  for (int k8 = 0; k8 < H / 8; ++k8) {
    uint4 u = rp[k8];
    const __bf16* bp = (const __bf16*)&u;
#pragma unroll
    for (int j = 0; j < 8; ++j) acc = fmaf((float)bp[j], gw2[k8 * 8 + j], acc);
  }
  gate[nd] = acc;
  int b = batch[nd];
  int iv = __float_as_int(acc);
  unsigned enc = (iv < 0) ? ~(unsigned)iv : ((unsigned)iv | 0x80000000u);
  atomicMax(&gmaxe[b], enc);
  atomicAdd(&counts[b], 1.f);
}

// eg = exp(gate - gmax[b]) ; denom[b] += eg
__global__ __launch_bounds__(256) void k_eg(
    const float* __restrict__ gate, const int* __restrict__ batch,
    const unsigned* __restrict__ gmaxe, float* __restrict__ egv,
    float* __restrict__ denom, int N)
{
  int nd = blockIdx.x * blockDim.x + threadIdx.x;
  if (nd >= N) return;
  int b = batch[nd];
  unsigned u = gmaxe[b];
  int iv = (u & 0x80000000u) ? (int)(u & 0x7fffffffu) : ~(int)u;
  float gm = __int_as_float(iv);
  float e = expf(gate[nd] - gm);
  egv[nd] = e;
  atomicAdd(&denom[b], e);
}

// attn[b] += (eg/denom[b]) * h
__global__ __launch_bounds__(256) void k_attn(
    const float* __restrict__ h, const float* __restrict__ egv, const float* __restrict__ denom,
    const int* __restrict__ batch, float* __restrict__ attn, int N)
{
  int idx = blockIdx.x * blockDim.x + threadIdx.x;
  if (idx >= N * H) return;
  int nd = idx >> 8, n = idx & 255;
  int b = batch[nd];
  float alpha = egv[nd] / fmaxf(denom[b], 1e-16f);
  atomicAdd(&attn[(size_t)b * H + n], alpha * h[idx]);
}

// out[g,t] = concat(mean,sums,mx,attn) @ cls_w + cls_b   (K=1024, T=12)
__global__ __launch_bounds__(256) void k_cls(
    const float* __restrict__ sums, const float* __restrict__ mx, const float* __restrict__ attn,
    const float* __restrict__ counts, const float* __restrict__ cw, const float* __restrict__ cb,
    float* __restrict__ out, int G, int T)
{
  int idx = blockIdx.x * blockDim.x + threadIdx.x;
  if (idx >= G * T) return;
  int g = idx / T, t = idx - g * T;
  float cnt = fmaxf(counts[g], 1.f);
  float acc = cb[t];
  for (int n = 0; n < H; ++n) {
    float s = sums[(size_t)g * H + n];
    acc = fmaf(s / cnt, cw[(size_t)(0 * H + n) * T + t], acc);
    acc = fmaf(s,       cw[(size_t)(1 * H + n) * T + t], acc);
    acc = fmaf(mx[(size_t)g * H + n],   cw[(size_t)(2 * H + n) * T + t], acc);
    acc = fmaf(attn[(size_t)g * H + n], cw[(size_t)(3 * H + n) * T + t], acc);
  }
  out[idx] = acc;
}

// ---------------------------------------------------------------------------
extern "C" void kernel_launch(void* const* d_in, const int* in_sizes, int n_in,
                              void* d_out, int out_size, void* d_ws, size_t ws_size,
                              hipStream_t stream) {
  (void)n_in; (void)ws_size;
  const float* x         = (const float*)d_in[0];
  const float* edge_attr = (const float*)d_in[1];
  const int*   eidx      = (const int*)  d_in[2];
  const int*   batch     = (const int*)  d_in[3];
  const float* em_w1 = (const float*)d_in[4];
  const float* em_b1 = (const float*)d_in[5];
  const float* em_w2 = (const float*)d_in[6];
  const float* em_b2 = (const float*)d_in[7];
  const float* c0_lin_w = (const float*)d_in[8];
  const float* c0_lin_b = (const float*)d_in[9];
  const float* c0_w1 = (const float*)d_in[10];
  const float* c0_b1 = (const float*)d_in[11];
  const float* c0_w2 = (const float*)d_in[12];
  const float* c0_b2 = (const float*)d_in[13];
  const float* bn0_g = (const float*)d_in[14];
  const float* bn0_b = (const float*)d_in[15];
  const float* lin_w = (const float*)d_in[16];
  const float* lin_b = (const float*)d_in[17];
  const float* w1 = (const float*)d_in[18];
  const float* b1 = (const float*)d_in[19];
  const float* w2 = (const float*)d_in[20];
  const float* b2 = (const float*)d_in[21];
  const float* bn_g = (const float*)d_in[22];
  const float* bn_b = (const float*)d_in[23];
  const float* gate_w1 = (const float*)d_in[24];
  const float* gate_b1 = (const float*)d_in[25];
  const float* gate_w2 = (const float*)d_in[26];
  const float* gate_b2 = (const float*)d_in[27];
  const float* cls_w = (const float*)d_in[28];
  const float* cls_b = (const float*)d_in[29];

  const int N = in_sizes[3];
  const int E = in_sizes[1] / 3;
  const int T = in_sizes[29];
  const int G = out_size / T;
  const int* src = eidx;
  const int* dst = eidx + E;

  // ---- workspace carve-out (256B aligned) ----
  char* p = (char*)d_ws;
  auto take = [&](size_t b) { char* r = p; p += (b + 255) & ~(size_t)255; return r; };
  __bf16* ea  = (__bf16*)take((size_t)E * H * 2);   // encoded edge feats (bf16, GEMM A)
  char*   uni = take((size_t)E * H * 2);            // union: ehid, then {h(f32), hb(bf16)}
  __bf16* ehid = (__bf16*)uni;
  float*  h    = (float*)uni;
  __bf16* hb   = (__bf16*)(uni + (size_t)N * H * 4);
  float*  agg  = (float*)take((size_t)N * H * 4);
  __bf16* An   = (__bf16*)take((size_t)N * H * 2);
  __bf16* T1   = (__bf16*)take((size_t)N * H * 2);
  float*  agg9 = (float*)take((size_t)N * 9 * 4);
  float*  gate = (float*)take((size_t)N * 4);
  float*  egv  = (float*)take((size_t)N * 4);
  float*  sums = (float*)take((size_t)G * H * 4);
  float*  mx   = (float*)take((size_t)G * H * 4);
  float*  attn = (float*)take((size_t)G * H * 4);
  float*  counts = (float*)take((size_t)G * 4);
  float*  denom  = (float*)take((size_t)G * 4);
  unsigned* gmaxe = (unsigned*)take((size_t)G * 4);
  __bf16* wT  = (__bf16*)take((size_t)15 * H * H * 2);
  __bf16* wT9 = (__bf16*)take((size_t)16 * H * 2);
  auto wTp = [&](int i) { return wT + (size_t)i * H * H; };

  // ---- weights -> bf16, transposed [N][K] for WMMA B fragments ----
  k_convT<<<H, H, 0, stream>>>(em_w2, wTp(0));
  k_convT<<<H, H, 0, stream>>>(c0_w2, wTp(1));
  for (int i = 0; i < 4; ++i) k_convT<<<H, H, 0, stream>>>(lin_w + (size_t)i * H * H, wTp(2 + i));
  for (int i = 0; i < 4; ++i) k_convT<<<H, H, 0, stream>>>(w1 + (size_t)i * H * H, wTp(6 + i));
  for (int i = 0; i < 4; ++i) k_convT<<<H, H, 0, stream>>>(w2 + (size_t)i * H * H, wTp(10 + i));
  k_convT<<<H, H, 0, stream>>>(gate_w1, wTp(14));
  k_convT9<<<16, H, 0, stream>>>(c0_lin_w, wT9);

  const int Et = E / 16, Nt = N / 16;          // both even
  const int NH = N * H;
  const int gemmBlkE = (Et + 15) / 16;         // 8 waves x 2 tiles = 32 rows... 16 tiles/block
  const int gemmBlkN = (Nt + 15) / 16;

  // MODE bits: 1=relu, 2=BN, 4=resid, 8=outF, 16=outB
  // ---- edge encoder: ea = relu(edge_attr@em_w1+b1) @ em_w2 + b2 (bf16) ----
  k_edge_hidden<<<E, 256, 0, stream>>>(edge_attr, em_w1, em_b1, ehid);
  k_gemm<16><<<gemmBlkE, 256, 0, stream>>>(ehid, wTp(0), em_b2,
      nullptr, nullptr, nullptr, nullptr, ea, Et);

  // ---- layer 0 (node dim 9, WMMA with N padded to 16) ----
  hipMemsetAsync(agg9, 0, (size_t)N * 9 * 4, stream);
  k_msg_scatter9<<<gemmBlkE, 256, 0, stream>>>(ea, wT9, c0_lin_b, x, src, dst, agg9, Et);
  k_c0_mlp1<<<(NH + 255) / 256, 256, 0, stream>>>(agg9, x, c0_w1, c0_b1, T1, N);
  k_gemm<1 | 2 | 8 | 16><<<gemmBlkN, 256, 0, stream>>>(T1, wTp(1), c0_b2,
      bn0_g, bn0_b, nullptr, h, hb, Nt);              // h = relu(bn0(...))

  // ---- layers 1..4 ----
  for (int i = 0; i < 4; ++i) {
    hipMemsetAsync(agg, 0, (size_t)N * H * 4, stream);
    k_msg_scatter<<<gemmBlkE, 256, 0, stream>>>(ea, wTp(2 + i), lin_b + i * H,
        h, src, dst, agg, Et);
    k_prep<<<(NH + 255) / 256, 256, 0, stream>>>(agg, h, An, N);
    k_gemm<1 | 16><<<gemmBlkN, 256, 0, stream>>>(An, wTp(6 + i), b1 + i * H,
        nullptr, nullptr, nullptr, nullptr, T1, Nt);              // relu -> bf16
    k_gemm<1 | 2 | 4 | 8 | 16><<<gemmBlkN, 256, 0, stream>>>(T1, wTp(10 + i), b2 + i * H,
        bn_g + i * H, bn_b + i * H, h, h, hb, Nt);                // h += relu(bn(...))
  }

  // ---- pooling ----
  hipMemsetAsync(sums, 0, (size_t)G * H * 4, stream);
  hipMemsetAsync(mx, 0, (size_t)G * H * 4, stream);
  hipMemsetAsync(attn, 0, (size_t)G * H * 4, stream);
  hipMemsetAsync(counts, 0, (size_t)G * 4, stream);
  hipMemsetAsync(denom, 0, (size_t)G * 4, stream);
  hipMemsetAsync(gmaxe, 0, (size_t)G * 4, stream);

  k_pool_nh<<<(NH + 255) / 256, 256, 0, stream>>>(h, batch, sums, mx, N);
  k_gemm<1 | 16><<<gemmBlkN, 256, 0, stream>>>(hb, wTp(14), gate_b1,
      nullptr, nullptr, nullptr, nullptr, T1, Nt);                // gate hidden (bf16)
  k_gate2<<<(N + 255) / 256, 256, 0, stream>>>(T1, gate_w2, gate_b2, batch, gate, gmaxe, counts, N);
  k_eg<<<(N + 255) / 256, 256, 0, stream>>>(gate, batch, gmaxe, egv, denom, N);
  k_attn<<<(NH + 255) / 256, 256, 0, stream>>>(h, egv, denom, batch, attn, N);
  k_cls<<<(G * T + 255) / 256, 256, 0, stream>>>(sums, mx, attn, counts, cls_w, cls_b,
      (float*)d_out, G, T);
}
